// SlotAttentionExperimental_41712722379323
// MI455X (gfx1250) — matrix operand order repaired and checked by reading, compiled, not run on hardware
//
#include <hip/hip_runtime.h>

// ---------------------------------------------------------------------------
// Slot-Attention (B=16,N=4096,D=512,NS=16,H=2048,ITERS=3) for gfx1250.
// All state fp32; every GEMM runs through v_wmma_f32_16x16x32_bf16 (bf16 in,
// fp32 accum). Weights (~24MB bf16) stay resident in the 192MB L2.
// GEMM: 256 threads = 8 waves (4M x 2N), wave tile 32x64 (2x4 WMMA frags,
// 8 WMMA per 12 b128 loads), block tile 128x128, WGP-scope prefetch.
// ---------------------------------------------------------------------------

#define DEV __device__ __forceinline__

typedef __attribute__((ext_vector_type(16))) __bf16 v16bf;
typedef __attribute__((ext_vector_type(8)))  float  v8f;

DEV unsigned short f2bf(float f) {
  unsigned int u = __float_as_uint(f);
  u += 0x7FFFu + ((u >> 16) & 1u);          // round-to-nearest-even
  return (unsigned short)(u >> 16);
}

union FragBF {
  v16bf v;
  uint4 q[2];
};

// ---------------------------------------------------------------------------
// Generic WMMA GEMM:  C[b] = alpha * A[b] @ W[b]^T + bias  (optional ReLU)
// A: [M,K] bf16 row-major (lda), W: [N,K] bf16 row-major (ldw, torch Linear
// layout), C: [M,N] fp32 (Cf) or bf16 (Cb), ldc. Batched via blockIdx.z.
//
// Fragment layouts (wave32, per CDNA5 ISA 7.12.2):
//  A 16x32 bf16: lane L (m=L&15, hi=L>>4) holds A[m][k0+8hi .. +7] and
//                A[m][k0+16+8hi .. +7]  -> two contiguous b128 loads.
//  B 32x16 bf16: lane L (n=L&15, hi=L>>4) holds W[n][k0+8hi .. +7] and
//                W[n][k0+16+8hi .. +7]  -> two b128 loads from one weight row.
//  C/D 16x16 f32: lane L (n=L&15, hi=L>>4), vgpr r holds row m = 8*hi + r.
// ---------------------------------------------------------------------------
__global__ __launch_bounds__(256) void k_gemm(
    const unsigned short* __restrict__ A, long long sA,
    const unsigned short* __restrict__ W, long long sW,
    const float* __restrict__ bias, float alpha,
    float* __restrict__ Cf, unsigned short* __restrict__ Cb, long long sC,
    int M, int N, int K, int lda, int ldw, int ldc, int relu)
{
  A += (long long)blockIdx.z * sA;
  W += (long long)blockIdx.z * sW;
  const long long cbase = (long long)blockIdx.z * sC;

  const int lane = threadIdx.x & 31;
  const int wave = threadIdx.x >> 5;
  const int m0 = blockIdx.x * 128 + (wave & 3)  * 32;   // 4 waves across M
  const int n0 = blockIdx.y * 128 + (wave >> 2) * 64;   // 2 waves across N
  const int lm = lane & 15;
  const int hi = lane >> 4;

  v8f acc[2][4];
  #pragma unroll
  for (int i = 0; i < 2; ++i)
    #pragma unroll
    for (int j = 0; j < 4; ++j)
      #pragma unroll
      for (int e = 0; e < 8; ++e) acc[i][j][e] = 0.f;

  // clamp OOB rows to valid addresses; results discarded at store time
  const unsigned short* ap[2];
  #pragma unroll
  for (int t = 0; t < 2; ++t) {
    const int m = min(m0 + t * 16 + lm, M - 1);
    ap[t] = A + (long long)m * lda + 8 * hi;
  }
  const unsigned short* bp[4];
  #pragma unroll
  for (int t = 0; t < 4; ++t) {
    const int n = min(n0 + t * 16 + lm, N - 1);
    bp[t] = W + (long long)n * ldw + 8 * hi;
  }

  #pragma unroll 2
  for (int k0 = 0; k0 < K; k0 += 32) {
    FragBF a[2], b[4];
    #pragma unroll
    for (int t = 0; t < 2; ++t) {
      a[t].q[0] = *(const uint4*)(ap[t] + k0);
      a[t].q[1] = *(const uint4*)(ap[t] + k0 + 16);
    }
    #pragma unroll
    for (int t = 0; t < 4; ++t) {
      b[t].q[0] = *(const uint4*)(bp[t] + k0);
      b[t].q[1] = *(const uint4*)(bp[t] + k0 + 16);
    }
    // WGP-scope prefetch of next K step (fills near caches; OOB is dropped)
    __builtin_prefetch(ap[0] + k0 + 32, 0, 3);
    __builtin_prefetch(ap[1] + k0 + 32, 0, 3);
    __builtin_prefetch(bp[0] + k0 + 32, 0, 3);
    __builtin_prefetch(bp[1] + k0 + 32, 0, 3);
    __builtin_prefetch(bp[2] + k0 + 32, 0, 3);
    __builtin_prefetch(bp[3] + k0 + 32, 0, 3);

    #pragma unroll
    for (int tm = 0; tm < 2; ++tm)
      #pragma unroll
      for (int tn = 0; tn < 4; ++tn)
        acc[tm][tn] = __builtin_amdgcn_wmma_f32_16x16x32_bf16(
            false, a[tm].v, false, b[tn].v, (short)0, acc[tm][tn], false, false);
  }

  #pragma unroll
  for (int tm = 0; tm < 2; ++tm) {
    #pragma unroll
    for (int tn = 0; tn < 4; ++tn) {
      const int col = n0 + tn * 16 + lm;
      if (col >= N) continue;
      const float bv = bias ? bias[col] : 0.f;
      #pragma unroll
      for (int r = 0; r < 8; ++r) {
        const int row = m0 + tm * 16 + 8 * hi + r;
        if (row >= M) continue;
        float o = alpha * acc[tm][tn][r] + bv;
        if (relu) o = fmaxf(o, 0.f);
        const long long off = cbase + (long long)row * ldc + col;
        if (Cf) Cf[off] = o; else Cb[off] = f2bf(o);
      }
    }
  }
}

// ---------------------------------------------------------------------------
// LayerNorm over D=512, one block (256 thr, 2 elems/thr) per row.
// Writes fp32 (of) and/or bf16 (ob).
// ---------------------------------------------------------------------------
__global__ __launch_bounds__(256) void k_layernorm(
    const float* __restrict__ x, const float* __restrict__ gw,
    const float* __restrict__ bw, float* __restrict__ of,
    unsigned short* __restrict__ ob)
{
  __shared__ float sm[16];
  const long long row = blockIdx.x;
  const int t = threadIdx.x;
  const float2 v = ((const float2*)(x + row * 512))[t];
  float s  = v.x + v.y;
  float sq = v.x * v.x + v.y * v.y;
  #pragma unroll
  for (int o = 16; o > 0; o >>= 1) { s += __shfl_down(s, o); sq += __shfl_down(sq, o); }
  if ((t & 31) == 0) { sm[t >> 5] = s; sm[8 + (t >> 5)] = sq; }
  __syncthreads();
  float S = 0.f, SQ = 0.f;
  #pragma unroll
  for (int w = 0; w < 8; ++w) { S += sm[w]; SQ += sm[8 + w]; }
  const float mean = S * (1.f / 512.f);
  const float var  = SQ * (1.f / 512.f) - mean * mean;
  const float rstd = rsqrtf(var + 1e-5f);
  const float o0 = (v.x - mean) * rstd * gw[2 * t]     + bw[2 * t];
  const float o1 = (v.y - mean) * rstd * gw[2 * t + 1] + bw[2 * t + 1];
  if (of) ((float2*)(of + row * 512))[t] = make_float2(o0, o1);
  if (ob) {
    ob[row * 512 + 2 * t]     = f2bf(o0);
    ob[row * 512 + 2 * t + 1] = f2bf(o1);
  }
}

// fp32 -> bf16 convert
__global__ __launch_bounds__(256) void k_f2bf(
    const float* __restrict__ s, unsigned short* __restrict__ d, long long n)
{
  const long long i = (long long)blockIdx.x * 256 + threadIdx.x;
  if (i < n) d[i] = f2bf(s[i]);
}

// slots = mu + exp(logsigma) * noise
__global__ __launch_bounds__(256) void k_slots_init(
    const float* __restrict__ noise, const float* __restrict__ mu,
    const float* __restrict__ ls, float* __restrict__ out)
{
  const int idx = blockIdx.x * 256 + threadIdx.x;
  if (idx >= 16 * 16 * 512) return;
  const int d = idx & 511;
  out[idx] = mu[d] + __expf(ls[d]) * noise[idx];
}

// row softmax over 16 entries, + EPS
__global__ __launch_bounds__(256) void k_softmax16(float* __restrict__ d, long long rows)
{
  const long long r = (long long)blockIdx.x * 256 + threadIdx.x;
  if (r >= rows) return;
  float* p = d + r * 16;
  float v[16];
  #pragma unroll
  for (int j = 0; j < 16; ++j) v[j] = p[j];
  float m = v[0];
  #pragma unroll
  for (int j = 1; j < 16; ++j) m = fmaxf(m, v[j]);
  float s = 0.f;
  #pragma unroll
  for (int j = 0; j < 16; ++j) { v[j] = __expf(v[j] - m); s += v[j]; }
  const float inv = 1.f / s;
  #pragma unroll
  for (int j = 0; j < 16; ++j) p[j] = v[j] * inv + 1e-8f;
}

// cs[b,j] = sum_i attn[b,i,j]   (i over 4096)
__global__ __launch_bounds__(256) void k_colsum16(
    const float* __restrict__ attn, float* __restrict__ cs)
{
  __shared__ float sm[8];
  const int j = blockIdx.x, b = blockIdx.y, t = threadIdx.x;
  const float* p = attn + ((long long)b * 4096) * 16 + j;
  float s = 0.f;
  for (int i = t; i < 4096; i += 256) s += p[(long long)i * 16];
  #pragma unroll
  for (int o = 16; o > 0; o >>= 1) s += __shfl_down(s, o);
  if ((t & 31) == 0) sm[t >> 5] = s;
  __syncthreads();
  if (t == 0) {
    float tot = 0.f;
    #pragma unroll
    for (int w = 0; w < 8; ++w) tot += sm[w];
    cs[b * 16 + j] = tot;
  }
}

// attn /= cs (in place, fp32); optionally emit bf16 zero-padded [.,32] copy
__global__ __launch_bounds__(256) void k_attn_div(
    float* __restrict__ attn, const float* __restrict__ cs,
    unsigned short* __restrict__ pad)
{
  const long long idx = (long long)blockIdx.x * 256 + threadIdx.x;
  if (idx >= (long long)16 * 4096 * 16) return;
  const int j = (int)(idx & 15);
  const long long bi = idx >> 4;           // b*4096 + i
  const int b = (int)(bi >> 12);
  const float v = attn[idx] / cs[b * 16 + j];
  attn[idx] = v;
  if (pad) pad[bi * 32 + j] = f2bf(v);     // cols 16..31 stay zero
}

// v_t[b,d,j] = v[b,j,d] (bf16, j padded to 32 with zeros)
__global__ __launch_bounds__(256) void k_vt_pad(
    const float* __restrict__ v, unsigned short* __restrict__ vt)
{
  const long long idx = (long long)blockIdx.x * 256 + threadIdx.x;
  if (idx >= (long long)16 * 512 * 32) return;
  const int j = (int)(idx & 31);
  const long long bd = idx >> 5;
  const int d = (int)(bd & 511), b = (int)(bd >> 9);
  const float val = (j < 16) ? v[((long long)b * 16 + j) * 512 + d] : 0.f;
  vt[idx] = f2bf(val);
}

// out[b,i,d] = sum_j attnT[b,j,i] * v[b,j,d]   (tiny s2i output einsum)
__global__ __launch_bounds__(256) void k_s2i_out(
    const float* __restrict__ attnT, const float* __restrict__ v,
    unsigned short* __restrict__ outb)
{
  const int b = blockIdx.y;
  const int t = threadIdx.x;
  const int i = t >> 4, dd = t & 15;
  const int d = blockIdx.x * 16 + dd;
  const float* vb = v + ((long long)b * 4096) * 512 + d;
  const float* ab = attnT + ((long long)b * 4096) * 16 + i;
  float acc = 0.f;
  #pragma unroll 4
  for (int j = 0; j < 4096; ++j)
    acc += ab[(long long)j * 16] * vb[(long long)j * 512];
  outb[((long long)b * 16 + i) * 512 + d] = f2bf(acc);
}

// GRUCell pointwise: gi/gh are [rows,1536] (r,z,n order), h updated in place
__global__ __launch_bounds__(256) void k_gru(
    const float* __restrict__ gi, const float* __restrict__ gh,
    float* __restrict__ h, long long rows)
{
  const long long idx = (long long)blockIdx.x * 256 + threadIdx.x;
  if (idx >= rows * 512) return;
  const long long r = idx >> 9;
  const int d = (int)(idx & 511);
  const float* gir = gi + r * 1536;
  const float* ghr = gh + r * 1536;
  const float ir = gir[d], iz = gir[512 + d], in_ = gir[1024 + d];
  const float hr = ghr[d], hz = ghr[512 + d], hn  = ghr[1024 + d];
  const float rg = 1.f / (1.f + __expf(-(ir + hr)));
  const float z  = 1.f / (1.f + __expf(-(iz + hz)));
  const float n  = tanhf(in_ + rg * hn);
  h[idx] = (1.f - z) * n + z * h[idx];
}

// ---------------------------------------------------------------------------
// Host orchestration
// ---------------------------------------------------------------------------
enum {
  IN_INPUTS = 0, IN_NOISE, LNI_G, LNI_B, MU, LOGSIG,
  S2I_LNIN_G, S2I_LNIN_B, S2I_LNCTX_G, S2I_LNCTX_B,
  S2I_QW, S2I_QB, S2I_KW, S2I_KB, S2I_VW, S2I_VB,
  S2I_WIH, S2I_WHH, S2I_BIH, S2I_BHH,
  SFF_LN_G, SFF_LN_B, SFF_W1, SFF_B1, SFF_W2, SFF_B2,
  SFF_WIH, SFF_WHH, SFF_BIH, SFF_BHH,
  I2S_LNIN_G, I2S_LNIN_B, I2S_LNCTX_G, I2S_LNCTX_B,
  I2S_QW, I2S_QB, I2S_KW, I2S_KB, I2S_VW, I2S_VB,
  I2S_WIH, I2S_WHH, I2S_BIH, I2S_BHH,
  IFF_LN_G, IFF_LN_B, IFF_W1, IFF_B1, IFF_W2, IFF_B2,
  IFF_WIH, IFF_WHH, IFF_BIH, IFF_BHH, NUM_IN
};

enum {
  WB_S2I_Q = 0, WB_S2I_K, WB_S2I_V, WB_S2I_WIH, WB_S2I_WHH,
  WB_SFF_W1, WB_SFF_W2, WB_SFF_WIH, WB_SFF_WHH,
  WB_I2S_Q, WB_I2S_K, WB_I2S_V, WB_I2S_WIH, WB_I2S_WHH,
  WB_IFF_W1, WB_IFF_W2, WB_IFF_WIH, WB_IFF_WHH, NUM_WB
};

extern "C" void kernel_launch(void* const* d_in, const int* in_sizes, int n_in,
                              void* d_out, int out_size, void* d_ws, size_t ws_size,
                              hipStream_t stream)
{
  (void)in_sizes; (void)out_size; (void)ws_size;

  constexpr int  Bb = 16, Nn = 4096, NS = 16, Dd = 512, Hh = 2048, TD = 1536;
  constexpr long long R  = (long long)Bb * Nn;   // 65536 token rows
  constexpr long long RS = (long long)Bb * NS;   // 256 slot rows
  const float SCALE = 0.044194173824159216f;     // 512^-0.5

  const float* in[NUM_IN];
  for (int i = 0; i < NUM_IN && i < n_in; ++i) in[i] = (const float*)d_in[i];

  // ---- workspace arena (256B-aligned bump allocator) ----
  char* base = (char*)d_ws;
  size_t off = 0;
  auto alloc = [&](size_t bytes) -> void* {
    void* r = base + off;
    off += (bytes + 255) & ~(size_t)255;
    return r;
  };

  // bf16 weight copies (converted each call; ~12M elements, negligible)
  static const int widx[NUM_WB] = {
    S2I_QW, S2I_KW, S2I_VW, S2I_WIH, S2I_WHH,
    SFF_W1, SFF_W2, SFF_WIH, SFF_WHH,
    I2S_QW, I2S_KW, I2S_VW, I2S_WIH, I2S_WHH,
    IFF_W1, IFF_W2, IFF_WIH, IFF_WHH };
  static const long long wele[NUM_WB] = {
    (long long)Dd*Dd, (long long)Dd*Dd, (long long)Dd*Dd, (long long)TD*Dd, (long long)TD*Dd,
    (long long)Hh*Dd, (long long)Dd*Hh, (long long)TD*Dd, (long long)TD*Dd,
    (long long)Dd*Dd, (long long)Dd*Dd, (long long)Dd*Dd, (long long)TD*Dd, (long long)TD*Dd,
    (long long)Hh*Dd, (long long)Dd*Hh, (long long)TD*Dd, (long long)TD*Dd };
  unsigned short* wb[NUM_WB];
  for (int i = 0; i < NUM_WB; ++i) wb[i] = (unsigned short*)alloc((size_t)wele[i] * 2);

  unsigned short* ABF  = (unsigned short*)alloc((size_t)R * Dd * 2);   // LN(x) bf16
  unsigned short* BBF  = (unsigned short*)alloc((size_t)R * Dd * 2);   // big q/k bf16
  void*           R1   = alloc((size_t)R * Hh * 2);                    // h1 bf16 OR v_x fp32
  unsigned short* UBF  = (unsigned short*)alloc((size_t)R * Dd * 2);   // GRU updates bf16
  unsigned short* HBF  = (unsigned short*)alloc((size_t)R * Dd * 2);   // bf16(x state)
  float*          GI   = (float*)alloc((size_t)R * TD * 4);            // gi gates
  float*          GH   = (float*)alloc((size_t)R * TD * 4);            // gh gates
  float*          DOTS = (float*)alloc((size_t)Bb * Nn * NS * 4);      // [B,4096,16]
  unsigned short* APAD = (unsigned short*)alloc((size_t)Bb * Nn * 32 * 2);
  float*          CS   = (float*)alloc(256 * 4);
  unsigned short* SLN  = (unsigned short*)alloc((size_t)RS * Dd * 2);
  unsigned short* SQ   = (unsigned short*)alloc((size_t)RS * Dd * 2);
  unsigned short* SK   = (unsigned short*)alloc((size_t)RS * Dd * 2);
  unsigned short* SUPD = (unsigned short*)alloc((size_t)RS * Dd * 2);
  unsigned short* SHBF = (unsigned short*)alloc((size_t)RS * Dd * 2);
  unsigned short* SH1  = (unsigned short*)alloc((size_t)RS * Hh * 2);
  float*          SV32 = (float*)alloc((size_t)RS * Dd * 4);
  unsigned short* SVT  = (unsigned short*)alloc((size_t)Bb * Dd * 32 * 2);
  float*          SGI  = (float*)alloc((size_t)RS * TD * 4);
  float*          SGH  = (float*)alloc((size_t)RS * TD * 4);

  // persistent state lives directly in d_out: [slots | x]
  float* slots = (float*)d_out;
  float* x     = slots + RS * Dd;

  auto gemm = [&](const unsigned short* A, long long sA,
                  const unsigned short* Wt, long long sW,
                  const float* bias, float alpha,
                  float* Cf, unsigned short* Cb, long long sC,
                  int M, int Nc, int K, int lda, int ldw, int ldc,
                  int relu, int batch) {
    dim3 g((unsigned)((M + 127) / 128), (unsigned)((Nc + 127) / 128), (unsigned)batch);
    k_gemm<<<g, 256, 0, stream>>>(A, sA, Wt, sW, bias, alpha, Cf, Cb, sC,
                                  M, Nc, K, lda, ldw, ldc, relu);
  };
  auto LN = [&](const float* src, long long rows, const float* g, const float* b,
                float* of, unsigned short* ob) {
    k_layernorm<<<(unsigned)rows, 256, 0, stream>>>(src, g, b, of, ob);
  };
  auto CVT = [&](const float* s, unsigned short* d, long long n) {
    k_f2bf<<<(unsigned)((n + 255) / 256), 256, 0, stream>>>(s, d, n);
  };

  // ---- setup ----
  for (int i = 0; i < NUM_WB; ++i) CVT(in[widx[i]], wb[i], wele[i]);
  k_slots_init<<<(unsigned)((RS * Dd + 255) / 256), 256, 0, stream>>>(
      in[IN_NOISE], in[MU], in[LOGSIG], slots);
  LN(in[IN_INPUTS], R, in[LNI_G], in[LNI_B], x, nullptr);       // x = LN(inputs)
  hipMemsetAsync(APAD, 0, (size_t)Bb * Nn * 32 * 2, stream);    // pad cols stay 0

  for (int it = 0; it < 3; ++it) {
    // ================= s2i: slots attend over x =================
    LN(slots, RS, in[S2I_LNIN_G], in[S2I_LNIN_B], nullptr, SLN);
    LN(x, R, in[S2I_LNCTX_G], in[S2I_LNCTX_B], nullptr, ABF);
    gemm(SLN, 0, wb[WB_S2I_Q], 0, in[S2I_QB], 1.f, nullptr, SQ, 0, (int)RS, Dd, Dd, Dd, Dd, Dd, 0, 1);
    gemm(ABF, 0, wb[WB_S2I_K], 0, in[S2I_KB], 1.f, nullptr, BBF, 0, (int)R, Dd, Dd, Dd, Dd, Dd, 0, 1);
    gemm(ABF, 0, wb[WB_S2I_V], 0, in[S2I_VB], 1.f, (float*)R1, nullptr, 0, (int)R, Dd, Dd, Dd, Dd, Dd, 0, 1);
    // dotsT[b,j,i] = SCALE * k[b,j,:] . q[b,i,:]   -> [B,4096,16]
    gemm(BBF, (long long)Nn * Dd, SQ, (long long)NS * Dd, nullptr, SCALE,
         DOTS, nullptr, (long long)Nn * NS, Nn, NS, Dd, Dd, Dd, NS, 0, Bb);
    k_softmax16<<<(unsigned)((R + 255) / 256), 256, 0, stream>>>(DOTS, R);   // over slots i
    k_colsum16<<<dim3(16, Bb), 256, 0, stream>>>(DOTS, CS);                  // sum over j
    k_attn_div<<<(unsigned)((R * 16 + 255) / 256), 256, 0, stream>>>(DOTS, CS, nullptr);
    k_s2i_out<<<dim3(32, Bb), 256, 0, stream>>>(DOTS, (const float*)R1, SUPD);
    CVT(slots, SHBF, RS * Dd);
    gemm(SUPD, 0, wb[WB_S2I_WIH], 0, in[S2I_BIH], 1.f, SGI, nullptr, 0, (int)RS, TD, Dd, Dd, Dd, TD, 0, 1);
    gemm(SHBF, 0, wb[WB_S2I_WHH], 0, in[S2I_BHH], 1.f, SGH, nullptr, 0, (int)RS, TD, Dd, Dd, Dd, TD, 0, 1);
    k_gru<<<(unsigned)((RS * Dd + 255) / 256), 256, 0, stream>>>(SGI, SGH, slots, RS);

    // ================= sff: slot feed-forward =================
    LN(slots, RS, in[SFF_LN_G], in[SFF_LN_B], nullptr, SLN);
    gemm(SLN, 0, wb[WB_SFF_W1], 0, in[SFF_B1], 1.f, nullptr, SH1, 0, (int)RS, Hh, Dd, Dd, Dd, Hh, 1, 1);
    gemm(SH1, 0, wb[WB_SFF_W2], 0, in[SFF_B2], 1.f, nullptr, SUPD, 0, (int)RS, Dd, Hh, Hh, Hh, Dd, 0, 1);
    CVT(slots, SHBF, RS * Dd);
    gemm(SUPD, 0, wb[WB_SFF_WIH], 0, in[SFF_BIH], 1.f, SGI, nullptr, 0, (int)RS, TD, Dd, Dd, Dd, TD, 0, 1);
    gemm(SHBF, 0, wb[WB_SFF_WHH], 0, in[SFF_BHH], 1.f, SGH, nullptr, 0, (int)RS, TD, Dd, Dd, Dd, TD, 0, 1);
    k_gru<<<(unsigned)((RS * Dd + 255) / 256), 256, 0, stream>>>(SGI, SGH, slots, RS);

    // ================= i2s: x attends over slots =================
    LN(x, R, in[I2S_LNIN_G], in[I2S_LNIN_B], nullptr, ABF);
    LN(slots, RS, in[I2S_LNCTX_G], in[I2S_LNCTX_B], nullptr, SLN);
    gemm(ABF, 0, wb[WB_I2S_Q], 0, in[I2S_QB], 1.f, nullptr, BBF, 0, (int)R, Dd, Dd, Dd, Dd, Dd, 0, 1);
    gemm(SLN, 0, wb[WB_I2S_K], 0, in[I2S_KB], 1.f, nullptr, SK, 0, (int)RS, Dd, Dd, Dd, Dd, Dd, 0, 1);
    gemm(SLN, 0, wb[WB_I2S_V], 0, in[I2S_VB], 1.f, SV32, nullptr, 0, (int)RS, Dd, Dd, Dd, Dd, Dd, 0, 1);
    k_vt_pad<<<(unsigned)(((long long)Bb * Dd * 32 + 255) / 256), 256, 0, stream>>>(SV32, SVT);
    // dots[b,i,j] = SCALE * q[b,i,:] . k[b,j,:]   -> [B,4096,16]
    gemm(BBF, (long long)Nn * Dd, SK, (long long)NS * Dd, nullptr, SCALE,
         DOTS, nullptr, (long long)Nn * NS, Nn, NS, Dd, Dd, Dd, NS, 0, Bb);
    k_softmax16<<<(unsigned)((R + 255) / 256), 256, 0, stream>>>(DOTS, R);   // over slots j
    k_colsum16<<<dim3(16, Bb), 256, 0, stream>>>(DOTS, CS);                  // sum over i
    k_attn_div<<<(unsigned)((R * 16 + 255) / 256), 256, 0, stream>>>(DOTS, CS, APAD);
    // upd_x[b] = attn_pad[b] @ v_t[b]^T : M=4096,K=32,N=512 (zero-padded K)
    gemm(APAD, (long long)Nn * 32, SVT, (long long)Dd * 32, nullptr, 1.f,
         nullptr, UBF, (long long)Nn * Dd, Nn, Dd, 32, 32, 32, Dd, 0, Bb);
    CVT(x, HBF, R * Dd);
    gemm(UBF, 0, wb[WB_I2S_WIH], 0, in[I2S_BIH], 1.f, GI, nullptr, 0, (int)R, TD, Dd, Dd, Dd, TD, 0, 1);
    gemm(HBF, 0, wb[WB_I2S_WHH], 0, in[I2S_BHH], 1.f, GH, nullptr, 0, (int)R, TD, Dd, Dd, Dd, TD, 0, 1);
    k_gru<<<(unsigned)((R * Dd + 255) / 256), 256, 0, stream>>>(GI, GH, x, R);

    // ================= iff: x feed-forward =================
    LN(x, R, in[IFF_LN_G], in[IFF_LN_B], nullptr, ABF);
    gemm(ABF, 0, wb[WB_IFF_W1], 0, in[IFF_B1], 1.f, nullptr, (unsigned short*)R1, 0,
         (int)R, Hh, Dd, Dd, Dd, Hh, 1, 1);
    gemm((unsigned short*)R1, 0, wb[WB_IFF_W2], 0, in[IFF_B2], 1.f, nullptr, UBF, 0,
         (int)R, Dd, Hh, Hh, Hh, Dd, 0, 1);
    CVT(x, HBF, R * Dd);
    gemm(UBF, 0, wb[WB_IFF_WIH], 0, in[IFF_BIH], 1.f, GI, nullptr, 0, (int)R, TD, Dd, Dd, Dd, TD, 0, 1);
    gemm(HBF, 0, wb[WB_IFF_WHH], 0, in[IFF_BHH], 1.f, GH, nullptr, 0, (int)R, TD, Dd, Dd, Dd, TD, 0, 1);
    k_gru<<<(unsigned)((R * Dd + 255) / 256), 256, 0, stream>>>(GI, GH, x, R);
  }
  // final (slots, x) already in d_out
}
